// ClassicMHSAAdapter_80736795230978
// MI455X (gfx1250) — matrix operand order repaired and checked by reading, not compiled, not run on hardware
//
#include <hip/hip_runtime.h>
#include <hip/hip_bf16.h>

typedef __attribute__((ext_vector_type(16))) __bf16 v16bf;
typedef __attribute__((ext_vector_type(8)))  __bf16 v8bf;
typedef __attribute__((ext_vector_type(8)))  float  v8f;
typedef __attribute__((ext_vector_type(4)))  float  v4f;

union ABf { v16bf v; v8bf h[2]; };

#define WMMA_BF16(A, B, C) \
  __builtin_amdgcn_wmma_f32_16x16x32_bf16(false, (A), false, (B), (short)0, (C), false, false)

constexpr int Bc = 2;        // batch
constexpr int Cc = 256;      // channels
constexpr int Nn = 4096;     // H*W
constexpr int NH = 8;        // heads
constexpr int HD = 32;       // head dim
constexpr int OC = 768;      // 3*C
// (1/sqrt(32)) * log2(e) folded into Q so softmax uses exp2
constexpr float QSCALE = 0.25505653f;

__device__ __forceinline__ v8f zero8() {
    v8f r;
#pragma unroll
    for (int i = 0; i < 8; ++i) r[i] = 0.0f;
    return r;
}

// ---------------- prep: qkv_w (f32 row-major) -> bf16 ----------------
__global__ void k_cvt_w(const float* __restrict__ w, __bf16* __restrict__ wb) {
    int i = blockIdx.x * blockDim.x + threadIdx.x;
    if (i < OC * Cc) wb[i] = (__bf16)w[i];
}

// ---------------- prep: x (b,c,n) f32 -> xt (b,n,c) bf16 ----------------
__global__ void k_transpose_x(const float* __restrict__ x, __bf16* __restrict__ xt) {
    __shared__ float tile[32][33];
    int cb = blockIdx.x * 32, nb = blockIdx.y * 32, b = blockIdx.z;
    int tx = threadIdx.x, ty = threadIdx.y;
#pragma unroll
    for (int i = 0; i < 4; ++i) {
        int c = cb + ty + i * 8;
        tile[ty + i * 8][tx] = x[(size_t)(b * Cc + c) * Nn + nb + tx];
    }
    __syncthreads();
#pragma unroll
    for (int i = 0; i < 4; ++i) {
        int n = nb + ty + i * 8;
        xt[(size_t)(b * Nn + n) * Cc + cb + tx] = (__bf16)tile[tx][ty + i * 8];
    }
}

// ---------------- QKV projection GEMM ----------------
// Out[b,o,n] = sum_c W[o,c] * X[b,c,n] + bias[o]
// Wave tile: M=16 (o) x N=64 (n), K-loop over 256 in steps of 32.
__global__ void k_qkv_gemm(const __bf16* __restrict__ wb, const __bf16* __restrict__ xt,
                           const float* __restrict__ bias,
                           __bf16* __restrict__ Qh, __bf16* __restrict__ Kh,
                           __bf16* __restrict__ Vh) {
    int wave = threadIdx.x >> 5;
    int lane = threadIdx.x & 31;
    int ot4  = blockIdx.x % 12;
    int n64  = (blockIdx.x / 12) & 63;
    int b    = blockIdx.x / (12 * 64);
    int otile = (ot4 * 4 + wave) * 16;
    int nbase = n64 * 64;
    int hi   = lane >> 4;
    int m    = lane & 15;
    int koff = hi * 8;
    int cb16 = hi * 16;

    float bb[8];
    {
        const float4* bp = (const float4*)(bias + otile + hi * 8);
        float4 b0 = bp[0], b1 = bp[1];
        bb[0] = b0.x; bb[1] = b0.y; bb[2] = b0.z; bb[3] = b0.w;
        bb[4] = b1.x; bb[5] = b1.y; bb[6] = b1.z; bb[7] = b1.w;
    }

    v8f acc[4];
#pragma unroll
    for (int s = 0; s < 4; ++s) acc[s] = zero8();

    const __bf16* wrow  = wb + (size_t)(otile + m) * Cc;
    const __bf16* xbase = xt + (size_t)b * Nn * Cc;

    for (int kb = 0; kb < Cc; kb += 32) {
        ABf a;
        a.h[0] = *(const v8bf*)(wrow + kb + koff);
        a.h[1] = *(const v8bf*)(wrow + kb + 16 + koff);
#pragma unroll
        for (int s = 0; s < 4; ++s) {
            int n = nbase + s * 16 + m;
            const __bf16* xp = xbase + (size_t)n * Cc + kb + koff;
            ABf bf;
            bf.h[0] = *(const v8bf*)(xp);
            bf.h[1] = *(const v8bf*)(xp + 16);
            acc[s] = WMMA_BF16(a.v, bf.v, acc[s]);
        }
    }

    int mat    = otile >> 8;
    int ow     = otile & 255;
    int head   = ow >> 5;
    int cstart = (ow & 31) + hi * 8;

    if (mat == 0) {
#pragma unroll
        for (int s = 0; s < 4; ++s) {
            int n = nbase + s * 16 + m;
            v8bf q;
#pragma unroll
            for (int v = 0; v < 8; ++v) q[v] = (__bf16)((acc[s][v] + bb[v]) * QSCALE);
            *(v8bf*)(Qh + ((size_t)(b * NH + head) * Nn + n) * HD + cstart) = q;
        }
    } else if (mat == 1) {
#pragma unroll
        for (int s = 0; s < 4; ++s) {
            int n = nbase + s * 16 + m;
            v8bf k;
#pragma unroll
            for (int v = 0; v < 8; ++v) k[v] = (__bf16)(acc[s][v] + bb[v]);
            *(v8bf*)(Kh + ((size_t)(b * NH + head) * Nn + n) * HD + cstart) = k;
        }
    } else {
#pragma unroll
        for (int s = 0; s < 4; ++s) {
            int n = nbase + s * 16 + m;
#pragma unroll
            for (int v = 0; v < 8; ++v) {
                Vh[((size_t)(b * NH + head) * HD + cstart + v) * Nn + n] =
                    (__bf16)(acc[s][v] + bb[v]);
            }
        }
    }
}

// ---------------- flash attention (K/V shared via LDS, double-buffered) ----
// 128-thread block: 4 waves x 2 q-tiles = 128 queries.
// 64-key chunks staged in LDS once per block, consumed by all 8 q-tiles.
__global__ void __launch_bounds__(128) k_attn(const __bf16* __restrict__ Qh,
                                              const __bf16* __restrict__ Kh,
                                              const __bf16* __restrict__ Vh,
                                              float* __restrict__ out) {
    // padded pitches (40 / 72 elems) => conflict-free ds_load_b128 fragment reads
    __shared__ __align__(16) __bf16 skbuf[2][64 * 40];       // K: 64 keys x 32c
    __shared__ __align__(16) __bf16 svbuf[2][32 * 72];       // V: 32c x 64 keys
    __shared__ __align__(16) __bf16 spbuf[4][2][16 * 72];    // P: per wave / q-tile

    int tid  = threadIdx.x;
    int wave = tid >> 5;
    int lane = tid & 31;
    int qb = (blockIdx.x & 31) * 8;        // first q-tile of this block
    int h  = (blockIdx.x >> 5) & 7;
    int b  = blockIdx.x >> 8;
    int bh = b * NH + h;
    int hi   = lane >> 4;
    int m    = lane & 15;
    int koff = hi * 8;
    int kh16 = hi * 16;

    const __bf16* kbase = Kh + (size_t)bh * Nn * HD;
    const __bf16* vbase = Vh + (size_t)bh * HD * Nn;

    // Q A-fragments for this wave's two q-tiles (resident)
    ABf aq[2];
#pragma unroll
    for (int t = 0; t < 2; ++t) {
        const __bf16* qp = Qh + ((size_t)bh * Nn + (qb + wave * 2 + t) * 16 + m) * HD;
        aq[t].h[0] = *(const v8bf*)(qp + koff);
        aq[t].h[1] = *(const v8bf*)(qp + 16 + koff);
    }

    v8f o[2][2];
    float mrow[2][8], lrow[2][8];
#pragma unroll
    for (int t = 0; t < 2; ++t) {
        o[t][0] = zero8(); o[t][1] = zero8();
#pragma unroll
        for (int v = 0; v < 8; ++v) { mrow[t][v] = -1e30f; lrow[t][v] = 0.0f; }
    }

    // staging registers (double buffer): 2 x v8bf for K, 2 for V per thread
    v8bf gk[2], gv[2];
    int kkey[2], kcol[2], vc[2], vj[2];
#pragma unroll
    for (int i = 0; i < 2; ++i) {
        int flat = (tid + i * 128) * 8;
        kkey[i] = flat >> 5; kcol[i] = flat & 31;   // K: 64 x 32
        vc[i]   = flat >> 6; vj[i]   = flat & 63;   // V: 32 x 64
    }
#pragma unroll
    for (int i = 0; i < 2; ++i) {
        gk[i] = *(const v8bf*)(kbase + (size_t)kkey[i] * HD + kcol[i]);
        gv[i] = *(const v8bf*)(vbase + (size_t)vc[i] * Nn + vj[i]);
    }

    int buf = 0;
    for (int jc = 0; jc < Nn; jc += 64) {
        // commit staged chunk to LDS
#pragma unroll
        for (int i = 0; i < 2; ++i) {
            *(v8bf*)(&skbuf[buf][kkey[i] * 40 + kcol[i]]) = gk[i];
            *(v8bf*)(&svbuf[buf][vc[i] * 72 + vj[i]])     = gv[i];
        }
        __syncthreads();

        // preload next chunk into registers (overlaps with compute below)
        int jn = (jc + 64 < Nn) ? jc + 64 : jc;
#pragma unroll
        for (int i = 0; i < 2; ++i) {
            gk[i] = *(const v8bf*)(kbase + (size_t)(jn + kkey[i]) * HD + kcol[i]);
            gv[i] = *(const v8bf*)(vbase + (size_t)vc[i] * Nn + jn + vj[i]);
        }
        // L2 prefetch two chunks ahead (global_prefetch_b8)
        int jp = (jc + 128 < Nn) ? jc + 128 : jc;
        __builtin_prefetch((const void*)(kbase + (size_t)(jp + (tid >> 1)) * HD), 0, 1);
        __builtin_prefetch((const void*)(vbase + (size_t)(tid & 31) * Nn + jp), 0, 1);

        // K B-fragments (4 key-tiles of 16), shared by both q-tiles
        ABf bk[4];
#pragma unroll
        for (int kt = 0; kt < 4; ++kt) {
            const __bf16* kp = &skbuf[buf][(kt * 16 + m) * 40 + koff];
            bk[kt].h[0] = *(const v8bf*)(kp);
            bk[kt].h[1] = *(const v8bf*)(kp + 16);
        }

        // S + online softmax + P staging, per q-tile
#pragma unroll
        for (int t = 0; t < 2; ++t) {
            v8f s[4];
#pragma unroll
            for (int kt = 0; kt < 4; ++kt) s[kt] = WMMA_BF16(aq[t].v, bk[kt].v, zero8());

            float mc[8];
#pragma unroll
            for (int v = 0; v < 8; ++v)
                mc[v] = fmaxf(fmaxf(s[0][v], s[1][v]), fmaxf(s[2][v], s[3][v]));
#pragma unroll
            for (int mask = 1; mask <= 8; mask <<= 1)
#pragma unroll
                for (int v = 0; v < 8; ++v)
                    mc[v] = fmaxf(mc[v], __shfl_xor(mc[v], mask, 32));

            float alpha[8];
#pragma unroll
            for (int v = 0; v < 8; ++v) {
                float mn = fmaxf(mrow[t][v], mc[v]);
                alpha[v] = exp2f(mrow[t][v] - mn);
                mrow[t][v] = mn;
            }
#pragma unroll
            for (int kt = 0; kt < 4; ++kt)
#pragma unroll
                for (int v = 0; v < 8; ++v)
                    s[kt][v] = exp2f(s[kt][v] - mrow[t][v]);

            float rs[8];
#pragma unroll
            for (int v = 0; v < 8; ++v)
                rs[v] = (s[0][v] + s[1][v]) + (s[2][v] + s[3][v]);
#pragma unroll
            for (int mask = 1; mask <= 8; mask <<= 1)
#pragma unroll
                for (int v = 0; v < 8; ++v)
                    rs[v] += __shfl_xor(rs[v], mask, 32);
#pragma unroll
            for (int v = 0; v < 8; ++v) {
                lrow[t][v] = lrow[t][v] * alpha[v] + rs[v];
                o[t][0][v] *= alpha[v];
                o[t][1][v] *= alpha[v];
            }
            // P tile -> LDS (D-layout -> A-layout transform)
#pragma unroll
            for (int kt = 0; kt < 4; ++kt)
#pragma unroll
                for (int v = 0; v < 8; ++v)
                    spbuf[wave][t][(v + hi * 8) * 72 + kt * 16 + m] = (__bf16)s[kt][v];
        }
        __syncthreads();

        // V B-fragments [j-step u][c-frag f], shared by both q-tiles
        ABf bv[2][2];
#pragma unroll
        for (int u = 0; u < 2; ++u)
#pragma unroll
            for (int f = 0; f < 2; ++f) {
                const __bf16* vp = &svbuf[buf][(f * 16 + m) * 72 + u * 32 + koff];
                bv[u][f].h[0] = *(const v8bf*)(vp);
                bv[u][f].h[1] = *(const v8bf*)(vp + 16);
            }

#pragma unroll
        for (int t = 0; t < 2; ++t) {
            ABf pa[2];
#pragma unroll
            for (int u = 0; u < 2; ++u) {
                const __bf16* pp = &spbuf[wave][t][m * 72 + u * 32 + koff];
                pa[u].h[0] = *(const v8bf*)(pp);
                pa[u].h[1] = *(const v8bf*)(pp + 16);
            }
#pragma unroll
            for (int u = 0; u < 2; ++u)
#pragma unroll
                for (int f = 0; f < 2; ++f)
                    o[t][f] = WMMA_BF16(pa[u].v, bv[u][f].v, o[t][f]);
        }
        buf ^= 1;
    }

    // epilogue: out[b, h*32+c, n] = O[i,c] / l[i]
#pragma unroll
    for (int t = 0; t < 2; ++t) {
        int nout = (qb + wave * 2 + t) * 16 + hi * 8;
#pragma unroll
        for (int f = 0; f < 2; ++f) {
            float* op = out + ((size_t)b * Cc + h * HD + f * 16 + m) * Nn + nout;
            v4f r0, r1;
#pragma unroll
            for (int v = 0; v < 4; ++v) {
                r0[v] = o[t][f][v]     / lrow[t][v];
                r1[v] = o[t][f][4 + v] / lrow[t][4 + v];
            }
            *(v4f*)op = r0;
            *(v4f*)(op + 4) = r1;
        }
    }
}

extern "C" void kernel_launch(void* const* d_in, const int* in_sizes, int n_in,
                              void* d_out, int out_size, void* d_ws, size_t ws_size,
                              hipStream_t stream) {
    const float* x     = (const float*)d_in[0];
    const float* qkv_w = (const float*)d_in[1];
    const float* qkv_b = (const float*)d_in[2];
    float* out = (float*)d_out;

    char* ws = (char*)d_ws;
    __bf16* wb = (__bf16*)(ws);                 // 768*256*2       = 393,216 B
    __bf16* xt = (__bf16*)(ws + 393216);        // 2*4096*256*2    = 4,194,304 B
    __bf16* Qh = (__bf16*)(ws + 4587520);       // 2*8*4096*32*2   = 4,194,304 B
    __bf16* Kh = (__bf16*)(ws + 8781824);       // 4,194,304 B
    __bf16* Vh = (__bf16*)(ws + 12976128);      // 4,194,304 B  (end 17,170,432)

    k_cvt_w<<<(OC * Cc + 255) / 256, 256, 0, stream>>>(qkv_w, wb);

    dim3 tb(32, 8);
    dim3 tg(Cc / 32, Nn / 32, Bc);
    k_transpose_x<<<tg, tb, 0, stream>>>(x, xt);

    k_qkv_gemm<<<12 * 64 * Bc, 128, 0, stream>>>(wb, xt, qkv_b, Qh, Kh, Vh);

    // 512 blocks x 128 threads: each block owns 128 queries of one (b,h)
    k_attn<<<Bc * NH * (Nn / 128), 128, 0, stream>>>(Qh, Kh, Vh, out);
}
